// GlobalAttention_16767552323853
// MI455X (gfx1250) — matrix operand order repaired and checked by reading, not compiled, not run on hardware
//
#include <hip/hip_runtime.h>
#include <hip/hip_bf16.h>

#define D_MODEL 512
#define NHEADS  8
#define HD      64
#define BATCH   4
#define SEQT    8192
#define NW      1024          // SEQT / 8
#define NBW     32            // BATCH * 8
#define NTOK    (BATCH*SEQT)  // 32768

typedef __attribute__((ext_vector_type(16))) _Float16 v16h;
typedef __attribute__((ext_vector_type(8)))  float    v8f;

union Frag { v16h v; uint4 q[2]; };

__device__ __forceinline__ v8f wmma16(const Frag& a, const Frag& b, v8f c) {
  return __builtin_amdgcn_wmma_f32_16x16x32_f16(false, a.v, false, b.v, (short)0, c, false, false);
}

// ---- CDNA5 async global->LDS (ASYNCcnt) with synchronous fallback ----
#if defined(__has_builtin)
#if __has_builtin(__builtin_amdgcn_global_load_async_to_lds_b128)
#define HAVE_ASYNC_LDS 1
#endif
#endif
#ifndef HAVE_ASYNC_LDS
#define HAVE_ASYNC_LDS 0
#endif

typedef int aint4 __attribute__((vector_size(16)));
typedef __attribute__((address_space(1))) aint4* gptr4;
typedef __attribute__((address_space(3))) aint4* lptr4;

__device__ __forceinline__ void cp_b128(_Float16* lds, const _Float16* g) {
#if HAVE_ASYNC_LDS
  __builtin_amdgcn_global_load_async_to_lds_b128((gptr4)g, (lptr4)lds, 0, 0);
#else
  *(uint4*)lds = *(const uint4*)g;
#endif
}

__device__ __forceinline__ void wait_async() {
#if HAVE_ASYNC_LDS
#if __has_builtin(__builtin_amdgcn_s_wait_asynccnt)
  __builtin_amdgcn_s_wait_asynccnt(0);
#else
  asm volatile("s_wait_asynccnt 0x0" ::: "memory");
#endif
#endif
}

// ---------------- pre-pass kernels ----------------

__global__ void k_cvt(const float* __restrict__ src, _Float16* __restrict__ dst, int n) {
  for (int i = blockIdx.x * blockDim.x + threadIdx.x; i < n; i += gridDim.x * blockDim.x)
    dst[i] = (_Float16)src[i];
}

// W (512x512 row-major, [k][n]) -> f16 transposed [n][k]
__global__ void k_cvtT(const float* __restrict__ w, _Float16* __restrict__ wT) {
  int i = blockIdx.x * 256 + threadIdx.x;       // 262144 threads
  int k = i >> 9, n = i & 511;
  wT[(size_t)n * 512 + k] = (_Float16)w[i];
}

// additive mask per (bw, key m): 0 if valid else -1e30; if no key valid -> key 0 forced valid
__global__ void k_mask(const int* __restrict__ pm, float* __restrict__ mk) {
  __shared__ int anyv;
  int bw = blockIdx.x, b = bw >> 3, w = bw & 7;
  if (threadIdx.x == 0) anyv = 0;
  __syncthreads();
  int local = 0;
  for (int i = 0; i < 4; ++i) {
    int m = threadIdx.x * 4 + i;
    int t = m * 8 + w;
    int valid = pm[b * SEQT + t] != 0;
    local |= valid;
    mk[bw * NW + m] = valid ? 0.f : -1e30f;
  }
  if (local) atomicOr(&anyv, 1);
  __syncthreads();
  if (threadIdx.x == 0 && !anyv) mk[bw * NW + 0] = 0.f;
}

// ---------------- tiled WMMA GEMM, async double-buffered, fused RoPE ----------------
// C(128x64 per block) = A(Mx512,f16) @ BT^T (+bias). modes: 0/1 = Q/K (RoPE, ->[bw][h][n][64])
// 2 = V (->[bw][h][d][n] transposed)   3 = final out projection (f32, token layout)
__global__ __launch_bounds__(256) void k_gemm(
    const _Float16* __restrict__ Asrc, const _Float16* __restrict__ BT,
    const float* __restrict__ bias, const float* __restrict__ cosp,
    const float* __restrict__ sinp, _Float16* __restrict__ dsth,
    float* __restrict__ dstf, int mode)
{
  __shared__ alignas(32) _Float16 As[2][128][32];  // 16 KB
  __shared__ alignas(32) _Float16 Bs[2][64][32];   //  8 KB
  int tid = threadIdx.x, wave = tid >> 5, lane = tid & 31;
  int m0 = blockIdx.x * 128, n0 = blockIdx.y * 64;

  int ar = tid >> 1, akk = (tid & 1) * 16;
  int bn = tid >> 2, bkb = (tid & 3) * 8;
  const _Float16* gA = &Asrc[(size_t)(m0 + ar) * 512];
  const _Float16* gB = &BT[(size_t)(n0 + bn) * 512];

  auto stage = [&](int buf, int k0) {
    cp_b128(&As[buf][ar][akk],     gA + k0 + akk);
    cp_b128(&As[buf][ar][akk + 8], gA + k0 + akk + 8);
    cp_b128(&Bs[buf][bn][bkb],     gB + k0 + bkb);
  };

  stage(0, 0);
  wait_async();
  __syncthreads();

  v8f z = {0.f,0.f,0.f,0.f,0.f,0.f,0.f,0.f};
  v8f acc[4] = {z, z, z, z};
  int arow = (wave << 4) + (lane & 15);
  int abase = (lane >> 4) << 3;
  int bko = (lane >> 4) * 8;

  for (int k0 = 0; k0 < 512; k0 += 32) {
    int buf = (k0 >> 5) & 1;
    if (k0 + 32 < 512) stage(buf ^ 1, k0 + 32);   // prefetch next tile (async)

    Frag fa;
    fa.q[0] = *(const uint4*)&As[buf][arow][abase];
    fa.q[1] = *(const uint4*)&As[buf][arow][abase + 16];
#pragma unroll
    for (int tn = 0; tn < 4; ++tn) {
      Frag fb;
      int bnn = tn * 16 + (lane & 15);
      fb.q[0] = *(const uint4*)&Bs[buf][bnn][bko];
      fb.q[1] = *(const uint4*)&Bs[buf][bnn][bko + 16];
      acc[tn] = wmma16(fa, fb, acc[tn]);
    }
    wait_async();
    __syncthreads();
  }

  int cn = lane & 15;
  int rbase = m0 + (wave << 4) + ((lane >> 4) << 3);
  if (mode == 3) {
#pragma unroll
    for (int tn = 0; tn < 4; ++tn) {
      int c = n0 + tn * 16 + cn;
      float bv = bias[c];
#pragma unroll
      for (int v = 0; v < 8; ++v)
        dstf[(size_t)(rbase + v) * 512 + c] = acc[tn][v] + bv;
    }
  } else if (mode == 2) {                 // V -> transposed per head [bw][h][d][n]
    int h = n0 >> 6;
#pragma unroll
    for (int tn = 0; tn < 4; ++tn) {
      int d = tn * 16 + cn;
      float bv = bias[n0 + d];
#pragma unroll
      for (int v = 0; v < 8; ++v) {
        int rr = rbase + v, b = rr >> 13, t = rr & 8191;
        int nn = t >> 3, w = t & 7, bw = b * 8 + w;
        dsth[((size_t)(bw * 8 + h) * 64 + d) * 1024 + nn] = (_Float16)(acc[tn][v] + bv);
      }
    }
  } else {                                // Q/K with fused RoPE -> [bw][h][n][64]
    int h = n0 >> 6;
#pragma unroll
    for (int tn = 0; tn < 2; ++tn) {
      int d1 = tn * 16 + cn, d2 = d1 + 32;
      float b1 = bias[n0 + d1], b2 = bias[n0 + d2];
#pragma unroll
      for (int v = 0; v < 8; ++v) {
        int rr = rbase + v, b = rr >> 13, t = rr & 8191;
        float q1 = acc[tn][v] + b1, q2 = acc[tn + 2][v] + b2;
        float o1 = q1 * cosp[t * 64 + d1] - q2 * sinp[t * 64 + d1];
        float o2 = q2 * cosp[t * 64 + d2] + q1 * sinp[t * 64 + d2];
        int nn = t >> 3, w = t & 7, bw = b * 8 + w;
        size_t base = ((size_t)(bw * 8 + h) * 1024 + nn) * 64;
        dsth[base + d1] = (_Float16)o1;
        dsth[base + d2] = (_Float16)o2;
      }
    }
  }
}

// ---------------- flash attention: block = (bw,h) x 128 queries, 8 waves ----------------
__global__ __launch_bounds__(256) void k_attn(
    const _Float16* __restrict__ Qh, const _Float16* __restrict__ Kh,
    const _Float16* __restrict__ Vt, const float* __restrict__ mk,
    _Float16* __restrict__ Ob)
{
  __shared__ alignas(32) _Float16 Ks[2][32][64];   // 8 KB  (32 keys x 64 d)
  __shared__ alignas(32) _Float16 Vs[2][64][32];   // 8 KB  (64 d   x 32 keys)
  __shared__ alignas(32) _Float16 Ps[8][16][32];   // 8 KB  per-wave P staging
  __shared__ float Sm[2][32];
  int tid = threadIdx.x, wave = tid >> 5, lane = tid & 31;
  int bwh = blockIdx.y, bw = bwh >> 3, h = bwh & 7;
  int q0 = blockIdx.x * 128;
  size_t head = (size_t)bwh * NW * HD;     // same for Qh/Kh/Vt

  // preload Q A-fragments (d 0..31 and 32..63)
  int qrow = q0 + (wave << 4) + (lane & 15);
  int abase = (lane >> 4) << 4;
  const _Float16* qp = Qh + head + (size_t)qrow * 64;
  Frag aq0, aq1;
  aq0.q[0] = *(const uint4*)(qp + abase);       aq0.q[1] = *(const uint4*)(qp + abase + 8);
  aq1.q[0] = *(const uint4*)(qp + 32 + abase);  aq1.q[1] = *(const uint4*)(qp + 32 + abase + 8);

  int kkr = tid >> 3, kdd = (tid & 7) * 8;
  int vdd = tid >> 2, vmm = (tid & 3) * 8;
  const _Float16* gK = Kh + head;
  const _Float16* gV = Vt + head;
  auto stageKV = [&](int buf, int j) {
    cp_b128(&Ks[buf][kkr][kdd], gK + (size_t)(j + kkr) * 64 + kdd);
    cp_b128(&Vs[buf][vdd][vmm], gV + (size_t)vdd * 1024 + j + vmm);
    if (tid < 32) Sm[buf][tid] = mk[bw * NW + j + tid];
  };

  stageKV(0, 0);
  wait_async();
  __syncthreads();

  v8f z = {0.f,0.f,0.f,0.f,0.f,0.f,0.f,0.f};
  v8f O[4] = {z, z, z, z};
  float Mrow[8], Lrow[8];
#pragma unroll
  for (int v = 0; v < 8; ++v) { Mrow[v] = -1e38f; Lrow[v] = 0.f; }

  for (int j = 0; j < NW; j += 32) {
    int buf = (j >> 5) & 1;
    if (j + 32 < NW) stageKV(buf ^ 1, j + 32);   // prefetch next K/V chunk (async)

    // S(16x32) = Q @ K^T for this wave's 16 queries, scaled + masked
    v8f S[2];
#pragma unroll
    for (int kh = 0; kh < 2; ++kh) {
      int kl = kh * 16 + (lane & 15);
      int ko = (lane >> 4) * 16;
      Frag b0, b1;
      b0.q[0] = *(const uint4*)&Ks[buf][kl][ko];      b0.q[1] = *(const uint4*)&Ks[buf][kl][ko + 8];
      b1.q[0] = *(const uint4*)&Ks[buf][kl][32 + ko]; b1.q[1] = *(const uint4*)&Ks[buf][kl][32 + ko + 8];
      v8f s = z;
      s = wmma16(aq0, b0, s);
      s = wmma16(aq1, b1, s);
      S[kh] = s * 0.125f + Sm[buf][kh * 16 + (lane & 15)];
    }

    // online softmax (rows live across 16-lane N-groups of the C layout)
#pragma unroll
    for (int v = 0; v < 8; ++v) {
      float mx = fmaxf(S[0][v], S[1][v]);
#pragma unroll
      for (int off = 8; off; off >>= 1) mx = fmaxf(mx, __shfl_xor(mx, off, 32));
      float Mn = fmaxf(Mrow[v], mx);
      float corr = __expf(Mrow[v] - Mn);
      float p0 = __expf(S[0][v] - Mn);
      float p1 = __expf(S[1][v] - Mn);
      float sum = p0 + p1;
#pragma unroll
      for (int off = 8; off; off >>= 1) sum += __shfl_xor(sum, off, 32);
      Lrow[v] = Lrow[v] * corr + sum;
      Mrow[v] = Mn;
      O[0][v] *= corr; O[1][v] *= corr; O[2][v] *= corr; O[3][v] *= corr;
      int prow = v + ((lane >> 4) << 3);
      Ps[wave][prow][lane & 15]        = (_Float16)p0;
      Ps[wave][prow][16 + (lane & 15)] = (_Float16)p1;
    }
    asm volatile("s_wait_dscnt 0x0" ::: "memory");  // per-wave LDS RAW on Ps

    Frag pa;
    pa.q[0] = *(const uint4*)&Ps[wave][lane & 15][abase];
    pa.q[1] = *(const uint4*)&Ps[wave][lane & 15][abase + 8];
#pragma unroll
    for (int dt = 0; dt < 4; ++dt) {
      Frag fv;
      int dd = dt * 16 + (lane & 15);
      int ko = (lane >> 4) * 16;
      fv.q[0] = *(const uint4*)&Vs[buf][dd][ko];
      fv.q[1] = *(const uint4*)&Vs[buf][dd][ko + 8];
      O[dt] = wmma16(pa, fv, O[dt]);
    }
    wait_async();
    __syncthreads();
  }

  // normalize + scatter to token-row layout (f16) for the Wo GEMM
  int b = bw >> 3, w = bw & 7;
#pragma unroll
  for (int v = 0; v < 8; ++v) {
    float inv = 1.0f / Lrow[v];
    int q = q0 + (wave << 4) + v + ((lane >> 4) << 3);
    int r = b * SEQT + q * 8 + w;
    size_t rb = (size_t)r * 512 + h * 64 + (lane & 15);
    Ob[rb + 0]  = (_Float16)(O[0][v] * inv);
    Ob[rb + 16] = (_Float16)(O[1][v] * inv);
    Ob[rb + 32] = (_Float16)(O[2][v] * inv);
    Ob[rb + 48] = (_Float16)(O[3][v] * inv);
  }
}

// ---------------- host launcher ----------------
extern "C" void kernel_launch(void* const* d_in, const int* in_sizes, int n_in,
                              void* d_out, int out_size, void* d_ws, size_t ws_size,
                              hipStream_t stream) {
  (void)in_sizes; (void)n_in; (void)out_size; (void)ws_size;
  const float* x    = (const float*)d_in[0];
  const int*   pm   = (const int*)d_in[1];
  const float* cosp = (const float*)d_in[2];
  const float* sinp = (const float*)d_in[3];
  const float* Wq = (const float*)d_in[4];  const float* bq = (const float*)d_in[5];
  const float* Wk = (const float*)d_in[6];  const float* bk = (const float*)d_in[7];
  const float* Wv = (const float*)d_in[8];  const float* bv = (const float*)d_in[9];
  const float* Wo = (const float*)d_in[10]; const float* bo = (const float*)d_in[11];

  char* ws = (char*)d_ws;
  size_t off = 0;
  auto carve = [&](size_t bytes) {
    char* p = ws + off; off += (bytes + 255) & ~(size_t)255; return p;
  };
  _Float16* Xb  = (_Float16*)carve((size_t)NTOK * 512 * 2);
  _Float16* WqT = (_Float16*)carve(512 * 512 * 2);
  _Float16* WkT = (_Float16*)carve(512 * 512 * 2);
  _Float16* WvT = (_Float16*)carve(512 * 512 * 2);
  _Float16* WoT = (_Float16*)carve(512 * 512 * 2);
  _Float16* Qh  = (_Float16*)carve((size_t)NBW * NHEADS * NW * HD * 2);
  _Float16* Kh  = (_Float16*)carve((size_t)NBW * NHEADS * NW * HD * 2);
  _Float16* Vt  = (_Float16*)carve((size_t)NBW * NHEADS * HD * NW * 2);
  _Float16* Ob  = (_Float16*)carve((size_t)NTOK * 512 * 2);
  float*    mk  = (float*)carve((size_t)NBW * NW * 4);

  k_cvt <<<4096, 256, 0, stream>>>(x, Xb, NTOK * 512);
  k_cvtT<<<1024, 256, 0, stream>>>(Wq, WqT);
  k_cvtT<<<1024, 256, 0, stream>>>(Wk, WkT);
  k_cvtT<<<1024, 256, 0, stream>>>(Wv, WvT);
  k_cvtT<<<1024, 256, 0, stream>>>(Wo, WoT);
  k_mask<<<NBW, 256, 0, stream>>>(pm, mk);

  dim3 gg(NTOK / 128, 512 / 64);
  k_gemm<<<gg, 256, 0, stream>>>(Xb, WqT, bq, cosp, sinp, Qh, nullptr, 0);
  k_gemm<<<gg, 256, 0, stream>>>(Xb, WkT, bk, cosp, sinp, Kh, nullptr, 1);
  k_gemm<<<gg, 256, 0, stream>>>(Xb, WvT, bv, cosp, sinp, Vt, nullptr, 2);
  k_attn<<<dim3(NW / 128, NBW * NHEADS), 256, 0, stream>>>(Qh, Kh, Vt, mk, Ob);
  k_gemm<<<gg, 256, 0, stream>>>(Ob, WoT, bo, cosp, sinp, nullptr, (float*)d_out, 3);
}
